// FlowModel_binder_25211458027674
// MI455X (gfx1250) — matrix operand order repaired and checked by reading, not compiled, not run on hardware
//
#include <hip/hip_runtime.h>
#include <math.h>

// ---------------- problem constants ----------------
constexpr int BB = 2;
constexpr int NN = 2048;
constexpr int KN = 30;
constexpr int DN = 256;
constexpr int DE = 128;
constexpr int NF = 12;
constexpr int EF = 28;
constexpr int MIW = 2 * DN + DE;   // 640
constexpr float EPS = 1e-6f;

// padded LDS strides (halves): row stride in dwords % 64 != 0, 16B aligned
constexpr int MI_STR = 648;   // 640+8   -> 1296 B/row
constexpr int T1_STR = 264;   // 256+8   ->  528 B/row
constexpr int T2_STR = 136;   // 128+8   ->  272 B/row

// ---------------- LDS layout for k_layer (bytes) ----------------
constexpr int SM_MI   = 0;                               // 30*648*2 = 38880
constexpr int SM_T1   = 38880;                           // 30*264*2 = 15840
constexpr int SM_T2   = 54720;                           // 30*136*2 =  8160
constexpr int SM_MSUM = 62880;                           // 2*256*4  =  2048
constexpr int SM_IDX  = 64928;                           // 32*4     =   128
constexpr int SM_MROW = 65056;                           // 32*4     =   128
constexpr int SM_TOT  = 65184;

// ---------------- vector types / unions ----------------
typedef __attribute__((ext_vector_type(16))) _Float16 v16h;
typedef __attribute__((ext_vector_type(8)))  _Float16 v8h;
typedef __attribute__((ext_vector_type(8)))  float    v8f;

union V16H { v16h v; v8h q[2]; _Float16 h[16]; };
union V8F  { v8f v; float f[8]; };

__device__ inline float softplus_f(float x) {
    return fmaxf(x, 0.f) + log1pf(expf(-fabsf(x)));
}

// =====================================================================
// Kernel 1: frames, Xc, node features -> node_h init
// =====================================================================
__global__ __launch_bounds__(256)
void k_node_init(const float* __restrict__ X, const int* __restrict__ C,
                 const float* __restrict__ Wn, const float* __restrict__ bn,
                 float* __restrict__ node_out, float* __restrict__ Rg,
                 float* __restrict__ Xcg, float* __restrict__ mask_out)
{
    int blk = blockIdx.x;
    int b = blk / NN, n = blk % NN;
    size_t base = (size_t)b * NN + n;
    __shared__ float feat[NF];
    __shared__ float maskS;

    if (threadIdx.x == 0) {
        const float* xp = X + base * 12;
        float Na[3] = { xp[0], xp[1], xp[2] };
        float CA[3] = { xp[3], xp[4], xp[5] };
        float Cc[3] = { xp[6], xp[7], xp[8] };
        float Oa[3] = { xp[9], xp[10], xp[11] };
        // u = nrm(C - CA)
        float u[3], v[3], w[3], t[3];
        float l = 0.f;
        for (int i = 0; i < 3; ++i) { t[i] = Cc[i] - CA[i]; l += t[i]*t[i]; }
        l = sqrtf(l) + EPS;
        for (int i = 0; i < 3; ++i) u[i] = t[i] / l;
        // v = nrm(v0 - (v0.u)u)
        float dot = 0.f;
        float v0[3];
        for (int i = 0; i < 3; ++i) { v0[i] = Na[i] - CA[i]; dot += v0[i]*u[i]; }
        l = 0.f;
        for (int i = 0; i < 3; ++i) { t[i] = v0[i] - dot*u[i]; l += t[i]*t[i]; }
        l = sqrtf(l) + EPS;
        for (int i = 0; i < 3; ++i) v[i] = t[i] / l;
        // w = cross(u,v)
        w[0] = u[1]*v[2] - u[2]*v[1];
        w[1] = u[2]*v[0] - u[0]*v[2];
        w[2] = u[0]*v[1] - u[1]*v[0];
        // R[d][c], columns = (u,v,w)
        float* Rp = Rg + base * 9;
        for (int d = 0; d < 3; ++d) { Rp[3*d+0] = u[d]; Rp[3*d+1] = v[d]; Rp[3*d+2] = w[d]; }
        // Xc
        float* Xp = Xcg + base * 3;
        for (int d = 0; d < 3; ++d)
            Xp[d] = 0.25f * (Na[d] + CA[d] + Cc[d] + Oa[d]);
        // node features: bond(CA-N), bond(C-CA), bond(O-C)
        const float* pairs[3][2] = { {CA, Na}, {Cc, CA}, {Oa, Cc} };
        for (int p = 0; p < 3; ++p) {
            float bv[3]; float nl = 0.f;
            for (int i = 0; i < 3; ++i) { bv[i] = pairs[p][0][i] - pairs[p][1][i]; nl += bv[i]*bv[i]; }
            nl = sqrtf(nl);
            for (int i = 0; i < 3; ++i) feat[p*4+i] = bv[i] / (nl + EPS);
            feat[p*4+3] = logf(nl + EPS);
        }
        maskS = (C[base] > 0) ? 1.f : 0.f;
        mask_out[base] = maskS;
    }
    __syncthreads();

    int c = threadIdx.x;     // 256 threads, one output channel each
    float s = bn[c];
    #pragma unroll
    for (int f = 0; f < NF; ++f) s += feat[f] * Wn[f * DN + c];
    node_out[base * DN + c] = s * maskS;
}

// =====================================================================
// Kernel 2: top-K neighbors by squared distance (iterative argmin)
// =====================================================================
__global__ __launch_bounds__(256)
void k_topk(const float* __restrict__ Xcg, const float* __restrict__ mask_i,
            int* __restrict__ idxW, float* __restrict__ o_idxf,
            float* __restrict__ mask_ij_out, float* __restrict__ denom_out)
{
    int blk = blockIdx.x;
    int b = blk / NN, n = blk % NN;
    size_t base = (size_t)b * NN + n;
    int tid = threadIdx.x;

    __shared__ float d2[NN];
    __shared__ float rv[256];
    __shared__ int   ri[256];
    __shared__ float denomS;

    float x0 = Xcg[base*3+0], x1 = Xcg[base*3+1], x2 = Xcg[base*3+2];
    for (int j = tid; j < NN; j += 256) {
        size_t jb = (size_t)b * NN + j;
        float dx = Xcg[jb*3+0] - x0;
        float dy = Xcg[jb*3+1] - x1;
        float dz = Xcg[jb*3+2] - x2;
        float d = dx*dx + dy*dy + dz*dz;
        d += 1.0e9f * (1.f - mask_i[jb]);
        if (j == n) d += 1.0e9f;
        d2[j] = d;
    }
    if (tid == 0) denomS = 0.f;
    __syncthreads();

    for (int k = 0; k < KN; ++k) {
        float bv = 1.0e30f; int bi = NN;
        for (int j = tid; j < NN; j += 256) {
            float v = d2[j];
            if (v < bv) { bv = v; bi = j; }
        }
        rv[tid] = bv; ri[tid] = bi;
        __syncthreads();
        for (int s = 128; s > 0; s >>= 1) {
            if (tid < s) {
                float ov = rv[tid+s]; int oi = ri[tid+s];
                if (ov < rv[tid] || (ov == rv[tid] && oi < ri[tid])) { rv[tid] = ov; ri[tid] = oi; }
            }
            __syncthreads();
        }
        if (tid == 0) {
            int best = ri[0];
            idxW[base*KN + k] = best;
            o_idxf[base*KN + k] = (float)best;
            float mj = mask_i[base] * mask_i[(size_t)b*NN + best];
            mask_ij_out[base*KN + k] = mj;
            denomS += mj;
            d2[best] = 4.0e9f;     // exclude from further picks
        }
        __syncthreads();
    }
    if (tid == 0) denom_out[base] = denomS + EPS;
}

// =====================================================================
// Kernel 3: edge features (rbf16 + dir3 + Rrel9) -> edge_h init
// =====================================================================
__global__ __launch_bounds__(128)
void k_edge_init(const float* __restrict__ Xcg, const float* __restrict__ Rg,
                 const int* __restrict__ idxW, const float* __restrict__ mask_ij,
                 const float* __restrict__ We, const float* __restrict__ be,
                 float* __restrict__ edge_out)
{
    int blk = blockIdx.x;
    int b = blk / NN, n = blk % NN;
    size_t base = (size_t)b * NN + n;
    int tid = threadIdx.x;
    __shared__ float feat[EF];

    for (int k = 0; k < KN; ++k) {
        if (tid == 0) {
            int j = idxW[base*KN + k];
            size_t jb = (size_t)b * NN + j;
            float dv[3];
            for (int d = 0; d < 3; ++d) dv[d] = Xcg[jb*3+d] - Xcg[base*3+d];
            float dist = sqrtf(dv[0]*dv[0] + dv[1]*dv[1] + dv[2]*dv[2]);
            const float* Rn = Rg + base * 9;
            const float* Rj = Rg + jb * 9;
            // rbf
            for (int i = 0; i < 16; ++i) {
                float c = 20.f * (float)i / 15.f;
                float z = (dist - c) / 1.25f;
                feat[i] = expf(-z*z);
            }
            // dirv = (R^T d)/(dist+eps)
            for (int c = 0; c < 3; ++c) {
                float loc = Rn[0*3+c]*dv[0] + Rn[1*3+c]*dv[1] + Rn[2*3+c]*dv[2];
                feat[16 + c] = loc / (dist + EPS);
            }
            // Rrel = R^T Rj
            for (int e = 0; e < 3; ++e)
                for (int f = 0; f < 3; ++f)
                    feat[19 + 3*e + f] = Rn[0*3+e]*Rj[0*3+f] + Rn[1*3+e]*Rj[1*3+f] + Rn[2*3+e]*Rj[2*3+f];
        }
        __syncthreads();
        float s = be[tid];
        #pragma unroll
        for (int f = 0; f < EF; ++f) s += feat[f] * We[f * DE + tid];
        edge_out[(base*KN + k) * DE + tid] = s * mask_ij[base*KN + k];
        __syncthreads();
    }
}

// =====================================================================
// Kernel: pack f32 weight matrix (KDIM x OUT) into WMMA B-fragment f16
// frag f = ntile*KC + kc ; within frag: lane*16 + j halves
// element = W[kc*32 + (lane>>4)*8 + (j&7) + (j>>3)*16][ntile*16 + (lane&15)]
// =====================================================================
__global__ __launch_bounds__(256)
void k_pack(const float* __restrict__ W, _Float16* __restrict__ out, int KDIM, int OUT)
{
    int total = KDIM * OUT;
    int tid = blockIdx.x * 256 + threadIdx.x;
    if (tid >= total) return;
    int f = tid >> 9;
    int r = tid & 511;
    int lane = r >> 4;
    int j = r & 15;
    int KC = KDIM >> 5;
    int ntile = f / KC;
    int kc = f - ntile * KC;
    int krow = kc * 32 + ((lane >> 4) << 3) + (j & 7) + ((j >> 3) << 4);
    int col = (ntile << 4) + (lane & 15);
    out[tid] = (_Float16)W[(size_t)krow * OUT + col];
}

// =====================================================================
// Software-pipelined dual-accumulator GEMM tile pair.
// One A-fragment (16x32 f16 from LDS) feeds two B-fragments (two ntiles),
// with next-chunk A/B prefetched before issuing the current WMMAs.
// A row is pre-clamped to <=29 by the caller (padded rows are discarded
// at the output stage, so clamping is safe and branch-free).
// =====================================================================
template<int KC>
__device__ inline void gemm_pair(const _Float16* __restrict__ abase,   // + row*stride + hb
                                 const _Float16* __restrict__ pB0,     // + lane*16
                                 const _Float16* __restrict__ pB1,     // + lane*16
                                 V8F& acc0, V8F& acc1)
{
    V16H a; v16h b0, b1;
    a.q[0] = *(const v8h*)(abase);
    a.q[1] = *(const v8h*)(abase + 16);
    b0 = *(const v16h*)(pB0);
    b1 = *(const v16h*)(pB1);
    #pragma unroll
    for (int kc = 0; kc < KC - 1; ++kc) {
        V16H an; v16h b0n, b1n;
        const _Float16* ap = abase + (kc + 1) * 32;
        an.q[0] = *(const v8h*)(ap);
        an.q[1] = *(const v8h*)(ap + 16);
        b0n = *(const v16h*)(pB0 + ((size_t)(kc + 1) << 9));
        b1n = *(const v16h*)(pB1 + ((size_t)(kc + 1) << 9));
        acc0.v = __builtin_amdgcn_wmma_f32_16x16x32_f16(false, a.v, false, b0, (short)0, acc0.v, false, false);
        acc1.v = __builtin_amdgcn_wmma_f32_16x16x32_f16(false, a.v, false, b1, (short)0, acc1.v, false, false);
        a = an; b0 = b0n; b1 = b1n;
    }
    acc0.v = __builtin_amdgcn_wmma_f32_16x16x32_f16(false, a.v, false, b0, (short)0, acc0.v, false, false);
    acc1.v = __builtin_amdgcn_wmma_f32_16x16x32_f16(false, a.v, false, b1, (short)0, acc1.v, false, false);
}

// =====================================================================
// Kernel 4: one message-passing layer, one workgroup (8 waves) per node
// =====================================================================
__global__ __launch_bounds__(256)
void k_layer(const float* __restrict__ node_in, float* __restrict__ node_out,
             float* __restrict__ edge_io,
             const int* __restrict__ idxW, const float* __restrict__ mask_ij,
             const float* __restrict__ mask_i, const float* __restrict__ denom,
             const _Float16* __restrict__ pW1, const float* __restrict__ b1,
             const _Float16* __restrict__ pW2, const float* __restrict__ b2,
             const _Float16* __restrict__ pU1, const float* __restrict__ bu1,
             const _Float16* __restrict__ pU2, const float* __restrict__ bu2)
{
    extern __shared__ char smem[];
    _Float16* mi   = (_Float16*)(smem + SM_MI);
    _Float16* t1   = (_Float16*)(smem + SM_T1);
    _Float16* t2   = (_Float16*)(smem + SM_T2);
    float*    msum = (float*)   (smem + SM_MSUM);   // [2][256] per-mtile column sums
    int*      idxs = (int*)     (smem + SM_IDX);
    float*    mrow = (float*)   (smem + SM_MROW);

    int blk = blockIdx.x;
    int b = blk / NN, n = blk % NN;
    size_t base = (size_t)b * NN + n;
    int tid = threadIdx.x;
    int lane = tid & 31;
    int wv = tid >> 5;

    if (tid < KN) idxs[tid] = idxW[base*KN + tid];
    if (tid < 32) mrow[tid] = (tid < KN) ? mask_ij[base*KN + tid] : 0.f;
    msum[tid] = 0.f;
    msum[tid + 256] = 0.f;
    __syncthreads();

    // ---- phase 0: build mi = [hi | hj | edge] as f16 in LDS ----
    const float* nin_i = node_in + base * DN;
    for (int t = tid; t < KN * MIW; t += 256) {
        int k = t / MIW;
        int c = t - k * MIW;
        float v;
        if (c < DN)            v = nin_i[c];
        else if (c < 2*DN)     v = node_in[((size_t)b*NN + idxs[k]) * DN + (c - DN)];
        else                   v = edge_io[(base*KN + k) * DE + (c - 2*DN)];
        mi[k * MI_STR + c] = (_Float16)v;
    }
    __syncthreads();

    int arow = lane & 15;            // A-fragment row within tile
    int hb = (lane >> 4) << 3;       // 0 or 8 (K half select)
    int col = lane & 15;
    int rsel = (lane < 16) ? 0 : 8;  // C/D row offset per lane half
    int laneoff = lane << 4;         // B-fragment lane offset (halves)

    // ---- phase 1: t1 = softplus(mi@Wm1+b1), t2 = softplus(mi@Wue1+bu1) ----
    // 24 pair-groups (16 m-path, 8 e-path), 3 per wave, wave-uniform.
    for (int g = 0; g < 3; ++g) {
        int G = wv * 3 + g;                   // 0..23
        bool epath = (G >= 16);
        int mtile, np;
        const _Float16* pB; const float* bias;
        if (!epath) { mtile = G >> 3; np = G & 7; pB = pW1; bias = b1; }
        else        { int Ge = G - 16; mtile = Ge >> 2; np = Ge & 3; pB = pU1; bias = bu1; }
        int nt0 = np * 2, nt1 = nt0 + 1;
        int row = mtile * 16 + arow;
        row = (row > KN-1) ? (KN-1) : row;    // clamp: padded rows discarded below
        V8F acc0, acc1;
        #pragma unroll
        for (int r = 0; r < 8; ++r) { acc0.f[r] = 0.f; acc1.f[r] = 0.f; }
        gemm_pair<20>(mi + row * MI_STR + hb,
                      pB + (((size_t)nt0 * 20) << 9) + laneoff,
                      pB + (((size_t)nt1 * 20) << 9) + laneoff,
                      acc0, acc1);
        _Float16* tout = epath ? t2 : t1;
        int tstr = epath ? T2_STR : T1_STR;
        int c0 = (nt0 << 4) + col, c1 = (nt1 << 4) + col;
        float bc0 = bias[c0], bc1 = bias[c1];
        #pragma unroll
        for (int r = 0; r < 8; ++r) {
            int rowk = mtile * 16 + rsel + r;
            if (rowk < KN) {
                tout[rowk * tstr + c0] = (_Float16)softplus_f(acc0.f[r] + bc0);
                tout[rowk * tstr + c1] = (_Float16)softplus_f(acc1.f[r] + bc1);
            }
        }
    }
    __syncthreads();

    // ---- phase 2: m = t1@Wm2+b2 (masked col-sums); e update = t2@Wue2+bu2 ----
    for (int g = 0; g < 3; ++g) {
        int G = wv * 3 + g;
        bool epath = (G >= 16);
        int mtile, np;
        const _Float16 *pB, *Asrc; const float* bias; int astr;
        if (!epath) { mtile = G >> 3; np = G & 7; pB = pW2; Asrc = t1; astr = T1_STR; bias = b2; }
        else        { int Ge = G - 16; mtile = Ge >> 2; np = Ge & 3; pB = pU2; Asrc = t2; astr = T2_STR; bias = bu2; }
        int nt0 = np * 2, nt1 = nt0 + 1;
        int row = mtile * 16 + arow;
        row = (row > KN-1) ? (KN-1) : row;
        V8F acc0, acc1;
        #pragma unroll
        for (int r = 0; r < 8; ++r) { acc0.f[r] = 0.f; acc1.f[r] = 0.f; }
        if (!epath) {
            gemm_pair<8>(Asrc + row * astr + hb,
                         pB + (((size_t)nt0 * 8) << 9) + laneoff,
                         pB + (((size_t)nt1 * 8) << 9) + laneoff,
                         acc0, acc1);
        } else {
            gemm_pair<4>(Asrc + row * astr + hb,
                         pB + (((size_t)nt0 * 4) << 9) + laneoff,
                         pB + (((size_t)nt1 * 4) << 9) + laneoff,
                         acc0, acc1);
        }
        int c0 = (nt0 << 4) + col, c1 = (nt1 << 4) + col;
        float bc0 = bias[c0], bc1 = bias[c1];
        if (!epath) {
            float s0 = 0.f, s1 = 0.f;
            #pragma unroll
            for (int r = 0; r < 8; ++r) {
                int rowk = mtile * 16 + rsel + r;
                float mw = mrow[rowk];                 // mrow[30..31]==0
                s0 += (acc0.f[r] + bc0) * mw;
                s1 += (acc1.f[r] + bc1) * mw;
            }
            s0 += __shfl_xor(s0, 16, 32);              // combine lane halves
            s1 += __shfl_xor(s1, 16, 32);
            if (lane < 16) {                           // unique writer per slot
                msum[mtile * 256 + c0] = s0;
                msum[mtile * 256 + c1] = s1;
            }
        } else {
            #pragma unroll
            for (int r = 0; r < 8; ++r) {
                int rowk = mtile * 16 + rsel + r;
                if (rowk < KN) {
                    size_t ea0 = (base*KN + rowk) * DE + c0;
                    size_t ea1 = (base*KN + rowk) * DE + c1;
                    float o0 = edge_io[ea0];
                    float o1 = edge_io[ea1];
                    edge_io[ea0] = (o0 + acc0.f[r] + bc0) * mrow[rowk];
                    edge_io[ea1] = (o1 + acc1.f[r] + bc1) * mrow[rowk];
                }
            }
        }
    }
    __syncthreads();

    // ---- phase 3: node update ----
    {
        int c = tid;
        float mi_ = mask_i[base];
        float dn = denom[base];
        float s = msum[c] + msum[c + 256];
        node_out[base * DN + c] = (node_in[base * DN + c] + s / dn) * mi_;
    }
}

// =====================================================================
// host launcher
// =====================================================================
extern "C" void kernel_launch(void* const* d_in, const int* in_sizes, int n_in,
                              void* d_out, int out_size, void* d_ws, size_t ws_size,
                              hipStream_t stream)
{
    (void)in_sizes; (void)n_in; (void)out_size; (void)ws_size;

    const float* X    = (const float*)d_in[0];
    const int*   C    = (const int*)  d_in[1];
    const float* Wn   = (const float*)d_in[2];
    const float* bn   = (const float*)d_in[3];
    const float* We   = (const float*)d_in[4];
    const float* be   = (const float*)d_in[5];
    const float* Wm1  = (const float*)d_in[6];
    const float* bm1  = (const float*)d_in[7];
    const float* Wm2  = (const float*)d_in[8];
    const float* bm2  = (const float*)d_in[9];
    const float* Wue1 = (const float*)d_in[10];
    const float* bue1 = (const float*)d_in[11];
    const float* Wue2 = (const float*)d_in[12];
    const float* bue2 = (const float*)d_in[13];

    // d_out layout (floats): node_h | edge_h | edge_idx | mask_i | mask_ij
    float* out = (float*)d_out;
    float* o_node   = out;                                  // 2*2048*256
    float* o_edge   = out + (size_t)BB*NN*DN;               // 2*2048*30*128
    float* o_idxf   = o_edge + (size_t)BB*NN*KN*DE;         // 2*2048*30
    float* o_maski  = o_idxf + (size_t)BB*NN*KN;            // 2*2048
    float* o_maskij = o_maski + (size_t)BB*NN;              // 2*2048*30

    // workspace layout
    char* ws = (char*)d_ws;
    float*    Rg     = (float*)(ws + 0);          // B*N*9*4   = 147456
    float*    Xcg    = (float*)(ws + 147456);     // B*N*3*4   =  49152
    float*    nodeA  = (float*)(ws + 196608);     // B*N*256*4 = 4194304
    float*    nodeB  = (float*)(ws + 4390912);    // B*N*256*4 = 4194304
    float*    denomW = (float*)(ws + 8585216);    // B*N*4     =  16384
    _Float16* pw     = (_Float16*)(ws + 8601600); // packed weights, 1966080 B
    int*      idxW   = (int*)(ws + 10567680);     // B*N*K*4   = 491520

    _Float16* pWm1 = pw;                                 // 3*640*256
    _Float16* pWm2 = pWm1 + (size_t)3*640*256;           // 3*256*256
    _Float16* pUe1 = pWm2 + (size_t)3*256*256;           // 3*640*128
    _Float16* pUe2 = pUe1 + (size_t)3*640*128;           // 3*128*128

    const int NB = BB * NN;   // 4096 node-blocks

    // pack weights into WMMA B-fragment order (f16)
    for (int l = 0; l < 3; ++l) {
        k_pack<<<(640*256 + 255)/256, 256, 0, stream>>>(Wm1  + (size_t)l*640*256, pWm1 + (size_t)l*640*256, 640, 256);
        k_pack<<<(256*256 + 255)/256, 256, 0, stream>>>(Wm2  + (size_t)l*256*256, pWm2 + (size_t)l*256*256, 256, 256);
        k_pack<<<(640*128 + 255)/256, 256, 0, stream>>>(Wue1 + (size_t)l*640*128, pUe1 + (size_t)l*640*128, 640, 128);
        k_pack<<<(128*128 + 255)/256, 256, 0, stream>>>(Wue2 + (size_t)l*128*128, pUe2 + (size_t)l*128*128, 128, 128);
    }

    k_node_init<<<NB, 256, 0, stream>>>(X, C, Wn, bn, nodeA, Rg, Xcg, o_maski);
    k_topk     <<<NB, 256, 0, stream>>>(Xcg, o_maski, idxW, o_idxf, o_maskij, denomW);
    k_edge_init<<<NB, 128, 0, stream>>>(Xcg, Rg, idxW, o_maskij, We, be, o_edge);

    // 3 message-passing layers (ping-pong; last layer writes d_out node region)
    const float* nin[3]  = { nodeA, nodeB, nodeA };
    float*       nout[3] = { nodeB, nodeA, o_node };
    for (int l = 0; l < 3; ++l) {
        k_layer<<<NB, 256, SM_TOT, stream>>>(
            nin[l], nout[l], o_edge, idxW, o_maskij, o_maski, denomW,
            pWm1 + (size_t)l*640*256, bm1 + (size_t)l*256,
            pWm2 + (size_t)l*256*256, bm2 + (size_t)l*256,
            pUe1 + (size_t)l*640*128, bue1 + (size_t)l*128,
            pUe2 + (size_t)l*128*128, bue2 + (size_t)l*128);
    }
}
